// EncoderPointsPointNetPP_65335042506979
// MI455X (gfx1250) — matrix operand order repaired and checked by reading, not compiled, not run on hardware
//
#include <hip/hip_runtime.h>
#include <hip/hip_bf16.h>

// PointNet++ forward for MI455X (gfx1250).
// Pipeline (B = 2048 clouds, N = 1024 pts, 6 ch):
//   k_fps_bq1 : FPS(32) + ball query(r=0.2, ns=32)           -> idx1, new_xyz1
//   k_sa1     : gather + MLP 6->64->64->128 + maxpool (WMMA) -> feats1 [B,32,128]
//   k_fps_bq2 : FPS(16) + ball query(r=0.4, ns=32)           -> idx2, new_xyz2
//   k_sa2     : gather + MLP 131->128->128->256 + maxpool    -> feats2 [B,16,256]
//   k_sa3     : MLP 259->256->512->768 + maxpool             -> g      [B,768]
//   k_fc      : batched GEMM [B,768]@[768,768]+b             -> out    [B,768]
// Matrix stages use V_WMMA_F32_16X16X32_F16 (f16 in, f32 acc); K padded to
// multiples of 32 with zeros on both A and B sides (exact math).
// Fragment loads are vectorized: A = 2x ds_load_b128 (contiguous per layout);
// B = 2x ds_load_b128 from LDS tiles pre-swizzled into fragment order.

typedef _Float16 half_t;
typedef __attribute__((ext_vector_type(16))) _Float16 v16h;
typedef __attribute__((ext_vector_type(8)))  _Float16 v8h;
typedef __attribute__((ext_vector_type(2)))  _Float16 v2h;
typedef __attribute__((ext_vector_type(8)))  float    v8f;

#define BN_SCALE 0.99999500003749969f
#define NPTS   1024
#define NCH    6

// ---------------------------------------------------------------------------
// WMMA fragment helpers (CDNA5 wave32 layouts, CDNA5 ISA §7.12.2)
// ---------------------------------------------------------------------------

__device__ inline v8f wmma32(v16h a, v16h b, v8f c) {
  // (neg_a, A, neg_b, B, c_mod, C, reuse_a, reuse_b)
  return __builtin_amdgcn_wmma_f32_16x16x32_f16(false, a, false, b,
                                                (short)0, c, false, false);
}

// A fragment: 16x32 f16, row-major source in LDS, lda (halfs) multiple of 8.
// lane: row = l&15 ; VGPR0-7 hold K = kb..kb+15,
// kb = (l&16)?16:0  ->  one contiguous 32B run per lane.
__device__ inline v16h load_a_lds(const half_t* __restrict__ base, int lda) {
  const int l   = threadIdx.x & 31;
  const int row = l & 15;
  const int kb  = (l & 16) ? 16 : 0;
  const half_t* p = base + row * lda + kb;
  const v8h lo = *(const v8h*)(p);        // K = kb .. kb+7
  const v8h hi = *(const v8h*)(p + 8);    // K = kb+8 .. kb+15
  return __builtin_shufflevector(lo, hi, 0, 1, 2, 3, 4, 5, 6, 7,
                                         8, 9, 10, 11, 12, 13, 14, 15);
}

// Swizzled B tiles: for tile (kt,nt) of B[K][N], lane l holds 16 contiguous
// halfs = column (nt*16 + (l&15)), rows kt*32 + klo .. klo+15, klo=(l&16)?16:0.
// Stored per (tile, lane) as 32 contiguous bytes -> fragment = 2x ds_load_b128.
__device__ inline v16h load_b_swz(const half_t* __restrict__ swz, int tile) {
  const int l = threadIdx.x & 31;
  const v8h* p = (const v8h*)(swz + ((tile << 5) + l) * 16);
  return __builtin_shufflevector(p[0], p[1], 0, 1, 2, 3, 4, 5, 6, 7,
                                             8, 9, 10, 11, 12, 13, 14, 15);
}

// Cooperative fill of swizzled B tiles from global f32 W[K][N] (kmax valid
// rows; zero-pad beyond). Paired writes (v2h). Assumes blockDim.x == 256.
__device__ inline void fill_b_swz(half_t* __restrict__ swz,
                                  const float* __restrict__ W,
                                  int N, int kmax, int KT, int NT) {
  const int total2 = KT * NT * 256;  // pairs of halfs
  for (int i2 = threadIdx.x; i2 < total2; i2 += 256) {
    const int i = i2 * 2;
    const int tile = i >> 9;
    const int rem  = i & 511;
    const int lane_ = rem >> 4, e = rem & 15;   // e even
    const int kt = tile / NT, nt = tile - kt * NT;
    const int col = lane_ & 15;
    const int klo = (lane_ & 16) ? 16 : 0;
    const int k = kt * 32 + klo + e;
    const int n = nt * 16 + col;
    v2h h;
    h[0] = (k     < kmax) ? (half_t)W[(size_t)k * N + n]       : (half_t)0.f;
    h[1] = (k + 1 < kmax) ? (half_t)W[(size_t)(k + 1) * N + n] : (half_t)0.f;
    ((v2h*)swz)[i2] = h;
  }
}

// B fragment streamed from global f32 weights (L2-hot), rows >= kmax are zero.
__device__ inline v16h load_b_gmem(const float* __restrict__ W, int ldb,
                                   int k0, int n0, int kmax) {
  const int l   = threadIdx.x & 31;
  const int col = n0 + (l & 15);
  const int kb  = k0 + ((l & 16) ? 16 : 0);
  if (kb + 32 < kmax)  // prefetch next K-slab -> global_prefetch_b8
    __builtin_prefetch(&W[(size_t)(kb + 32) * ldb + col], 0, 1);
  v16h b;
#pragma unroll
  for (int v = 0; v < 8; ++v) {
    const int k = kb + 2 * v;
    b[2 * v]     = (k     < kmax) ? (half_t)W[(size_t)k * ldb + col]       : (half_t)0.f;
    b[2 * v + 1] = (k + 1 < kmax) ? (half_t)W[(size_t)(k + 1) * ldb + col] : (half_t)0.f;
  }
  return b;
}

// Store 16x16 f32 accumulator tile to f16 activation in LDS with
// relu((acc + bias) * BN_SCALE). C/D layout: row = v + 8*(l>=16), col = l&15.
__device__ inline void store_act(half_t* __restrict__ act, int ldc, int m0,
                                 int n0, v8f acc, const float* __restrict__ bias) {
  const int l   = threadIdx.x & 31;
  const int col = n0 + (l & 15);
  const int r0  = m0 + ((l & 16) ? 8 : 0);
  const float bv = bias[col];
#pragma unroll
  for (int v = 0; v < 8; ++v) {
    float f = (acc[v] + bv) * BN_SCALE;
    act[(r0 + v) * ldc + col] = (half_t)fmaxf(f, 0.f);
  }
}

// Maxpool over 32 rows (two 16x16 tiles) with relu-bias, write 16 f32 outputs.
__device__ inline void maxpool32_store(float* __restrict__ out, int n0,
                                       v8f a0, v8f a1,
                                       const float* __restrict__ bias) {
  const int l   = threadIdx.x & 31;
  const int col = n0 + (l & 15);
  const float bv = bias[col];
  float m = 0.f;  // relu => outputs >= 0
#pragma unroll
  for (int v = 0; v < 8; ++v) {
    m = fmaxf(m, fmaxf(0.f, (a0[v] + bv) * BN_SCALE));
    m = fmaxf(m, fmaxf(0.f, (a1[v] + bv) * BN_SCALE));
  }
  m = fmaxf(m, __shfl_xor(m, 16, 32));  // merge rows v..v+7 with v+8..v+15
  if (l < 16) out[col] = m;
}

// Maxpool over 16 rows (one tile).
__device__ inline void maxpool16_store(float* __restrict__ out, int n0, v8f a0,
                                       const float* __restrict__ bias) {
  const int l   = threadIdx.x & 31;
  const int col = n0 + (l & 15);
  const float bv = bias[col];
  float m = 0.f;
#pragma unroll
  for (int v = 0; v < 8; ++v)
    m = fmaxf(m, fmaxf(0.f, (a0[v] + bv) * BN_SCALE));
  m = fmaxf(m, __shfl_xor(m, 16, 32));
  if (l < 16) out[col] = m;
}

// ---------------------------------------------------------------------------
// Kernel 1: FPS(32) + ball query (radius 0.2, nsample 32). One block / cloud.
// Point staging uses CDNA5 async global->LDS copies (ASYNCcnt tracked):
// two 8B-aligned b64 transfers per point into a stride-4 LDS layout.
// ---------------------------------------------------------------------------
__global__ void __launch_bounds__(256)
k_fps_bq1(const float* __restrict__ x, int* __restrict__ idx1,
          float* __restrict__ nxyz1) {
  __shared__ float sx[NPTS * 4];   // x,y,z,(ch3 ignored) per point
  __shared__ float dist[NPTS];
  __shared__ float rv[256];
  __shared__ int   ri[256];
  __shared__ int   centers[32];
  const int b = blockIdx.x, t = threadIdx.x;

  const uint32_t lds_sx = (uint32_t)(uintptr_t)(void*)sx;
  for (int i = t; i < NPTS; i += 256) {
    const unsigned long long gp =
        (unsigned long long)(uintptr_t)(x + ((size_t)b * NPTS + i) * NCH);
    const uint32_t l0 = lds_sx + (uint32_t)i * 16u;
    asm volatile("global_load_async_to_lds_b64 %0, %1, off"
                 :: "v"(l0), "v"(gp) : "memory");
    asm volatile("global_load_async_to_lds_b64 %0, %1, off"
                 :: "v"(l0 + 8u), "v"(gp + 8ull) : "memory");
  }
  for (int i = t; i < NPTS; i += 256) dist[i] = 1e10f;
  if (t == 0) centers[0] = 0;
  asm volatile("s_wait_asynccnt 0x0" ::: "memory");
  __syncthreads();

  int last = 0;
  for (int it = 1; it < 32; ++it) {
    const float px = sx[last * 4 + 0], py = sx[last * 4 + 1], pz = sx[last * 4 + 2];
    float bv = -1.f; int bi = 0;
    for (int i = t; i < NPTS; i += 256) {
      const float dx = sx[i * 4 + 0] - px, dy = sx[i * 4 + 1] - py,
                  dz = sx[i * 4 + 2] - pz;
      const float nd = fminf(dist[i], dx * dx + dy * dy + dz * dz);
      dist[i] = nd;
      if (nd > bv) { bv = nd; bi = i; }
    }
    rv[t] = bv; ri[t] = bi;
    __syncthreads();
    for (int s = 128; s > 0; s >>= 1) {
      if (t < s) {
        if (rv[t + s] > rv[t] || (rv[t + s] == rv[t] && ri[t + s] < ri[t])) {
          rv[t] = rv[t + s]; ri[t] = ri[t + s];
        }
      }
      __syncthreads();
    }
    last = ri[0];
    if (t == 0) centers[it] = last;
    __syncthreads();
  }

  if (t < 32) {
    const int c = centers[t];
    const float cx = sx[c * 4 + 0], cy = sx[c * 4 + 1], cz = sx[c * 4 + 2];
    float* np = nxyz1 + ((size_t)b * 32 + t) * 3;
    np[0] = cx; np[1] = cy; np[2] = cz;
    int* ip = idx1 + ((size_t)b * 32 + t) * 32;
    int cnt = 0, first = -1;
    for (int j = 0; j < NPTS; ++j) {
      const float dx = sx[j * 4 + 0] - cx, dy = sx[j * 4 + 1] - cy,
                  dz = sx[j * 4 + 2] - cz;
      if (dx * dx + dy * dy + dz * dz < 0.04f) {
        if (first < 0) first = j;
        if (cnt < 32) ip[cnt++] = j;
      }
    }
    if (first < 0) first = 0;
    for (; cnt < 32; ++cnt) ip[cnt] = first;
  }
}

// ---------------------------------------------------------------------------
// Kernel 2: SA1 MLP 6->64->64->128 + maxpool. One block / cloud, 8 waves.
// Each wave handles 4 centers (32 sample rows each). Weights in LDS f16,
// pre-swizzled into B-fragment order (2x ds_load_b128 per fragment).
// ---------------------------------------------------------------------------
__global__ void __launch_bounds__(256)
k_sa1(const float* __restrict__ x, const float* __restrict__ nxyz1,
      const int* __restrict__ idx1,
      const float* __restrict__ w0, const float* __restrict__ b0,
      const float* __restrict__ w1, const float* __restrict__ b1,
      const float* __restrict__ w2, const float* __restrict__ b2,
      float* __restrict__ feats1) {
  extern __shared__ char smem[];
  half_t* w0z = (half_t*)smem;          // KT=1,NT=4  :  4 tiles * 512
  half_t* w1z = w0z + 4 * 512;          // KT=2,NT=4  :  8 tiles * 512
  half_t* w2z = w1z + 8 * 512;          // KT=2,NT=8  : 16 tiles * 512
  half_t* a0  = w2z + 16 * 512;         // 8 waves x [32][32]
  half_t* ac1 = a0 + 8 * 32 * 32;       // 8 waves x [32][64]
  half_t* ac2 = ac1 + 8 * 32 * 64;      // 8 waves x [32][64]

  const int b = blockIdx.x, t = threadIdx.x;
  const int wv = t >> 5, lane = t & 31;

  fill_b_swz(w0z, w0, 64, 6, 1, 4);     // [6,64]   K-pad -> 32
  fill_b_swz(w1z, w1, 64, 64, 2, 4);    // [64,64]
  fill_b_swz(w2z, w2, 128, 64, 2, 8);   // [64,128]
  __syncthreads();

  half_t* myA = a0 + wv * 32 * 32;
  half_t* my1 = ac1 + wv * 32 * 64;
  half_t* my2 = ac2 + wv * 32 * 64;

  for (int ci = 0; ci < 4; ++ci) {
    const int c = wv + ci * 8;
    {  // gather A: row = lane (sample), 6 valid channels, pad K to 32
      const int j = idx1[((size_t)b * 32 + c) * 32 + lane];
      const float* pj = x + ((size_t)b * NPTS + j) * NCH;
      const float* pc = nxyz1 + ((size_t)b * 32 + c) * 3;
      half_t* r = myA + lane * 32;
      r[0] = (half_t)(pj[0] - pc[0]);
      r[1] = (half_t)(pj[1] - pc[1]);
      r[2] = (half_t)(pj[2] - pc[2]);
      r[3] = (half_t)pj[3]; r[4] = (half_t)pj[4]; r[5] = (half_t)pj[5];
#pragma unroll
      for (int k = 6; k < 32; ++k) r[k] = (half_t)0.f;
    }
    __syncthreads();

    // stage 1: [32,32(K=6)] @ [32,64]
#pragma unroll
    for (int mt = 0; mt < 2; ++mt)
#pragma unroll
      for (int nt = 0; nt < 4; ++nt) {
        v8f acc = {};
        acc = wmma32(load_a_lds(myA + mt * 16 * 32, 32),
                     load_b_swz(w0z, nt), acc);
        store_act(my1, 64, mt * 16, nt * 16, acc, b0);
      }
    __syncthreads();

    // stage 2: [32,64] @ [64,64]
#pragma unroll
    for (int mt = 0; mt < 2; ++mt)
#pragma unroll
      for (int nt = 0; nt < 4; ++nt) {
        v8f acc = {};
#pragma unroll
        for (int kt = 0; kt < 2; ++kt)
          acc = wmma32(load_a_lds(my1 + mt * 16 * 64 + kt * 32, 64),
                       load_b_swz(w1z, kt * 4 + nt), acc);
        store_act(my2, 64, mt * 16, nt * 16, acc, b1);
      }
    __syncthreads();

    // stage 3: [32,64] @ [64,128] fused with maxpool over the 32 samples
    float* outp = feats1 + ((size_t)b * 32 + c) * 128;
#pragma unroll
    for (int nt = 0; nt < 8; ++nt) {
      v8f acc0 = {}, acc1 = {};
#pragma unroll
      for (int kt = 0; kt < 2; ++kt) {
        v16h bb = load_b_swz(w2z, kt * 8 + nt);
        acc0 = wmma32(load_a_lds(my2 + kt * 32, 64), bb, acc0);
        acc1 = wmma32(load_a_lds(my2 + 16 * 64 + kt * 32, 64), bb, acc1);
      }
      maxpool32_store(outp, nt * 16, acc0, acc1, b2);
    }
    __syncthreads();
  }
}

// ---------------------------------------------------------------------------
// Kernel 3: FPS(16) + ball query (radius 0.4, ns 32) over 32 pts. 1 wave/cloud.
// ---------------------------------------------------------------------------
__global__ void __launch_bounds__(32)
k_fps_bq2(const float* __restrict__ nx1, int* __restrict__ idx2,
          float* __restrict__ nx2) {
  __shared__ float sx[32 * 3];
  __shared__ int centers[16];
  const int b = blockIdx.x, l = threadIdx.x;

  const float* p = nx1 + ((size_t)b * 32 + l) * 3;
  sx[l * 3 + 0] = p[0]; sx[l * 3 + 1] = p[1]; sx[l * 3 + 2] = p[2];
  if (l == 0) centers[0] = 0;
  __syncthreads();

  const float mx = sx[l * 3 + 0], my = sx[l * 3 + 1], mz = sx[l * 3 + 2];
  float dmin = 1e10f;
  int last = 0;
  for (int it = 1; it < 16; ++it) {
    const float px = sx[last * 3 + 0], py = sx[last * 3 + 1], pz = sx[last * 3 + 2];
    const float dx = mx - px, dy = my - py, dz = mz - pz;
    dmin = fminf(dmin, dx * dx + dy * dy + dz * dz);
    float bv = dmin; int bi = l;
#pragma unroll
    for (int off = 16; off > 0; off >>= 1) {
      const float ov = __shfl_xor(bv, off, 32);
      const int   oi = __shfl_xor(bi, off, 32);
      if (ov > bv || (ov == bv && oi < bi)) { bv = ov; bi = oi; }
    }
    last = bi;
    if (l == 0) centers[it] = last;
  }
  __syncthreads();

  if (l < 16) {
    const int c = centers[l];
    const float cx = sx[c * 3 + 0], cy = sx[c * 3 + 1], cz = sx[c * 3 + 2];
    float* np = nx2 + ((size_t)b * 16 + l) * 3;
    np[0] = cx; np[1] = cy; np[2] = cz;
    int* ip = idx2 + ((size_t)b * 16 + l) * 32;
    int cnt = 0, first = -1;
    for (int j = 0; j < 32; ++j) {
      const float dx = sx[j * 3 + 0] - cx, dy = sx[j * 3 + 1] - cy,
                  dz = sx[j * 3 + 2] - cz;
      if (dx * dx + dy * dy + dz * dz < 0.16f) {
        if (first < 0) first = j;
        if (cnt < 32) ip[cnt++] = j;
      }
    }
    if (first < 0) first = 0;
    for (; cnt < 32; ++cnt) ip[cnt] = first;
  }
}

// ---------------------------------------------------------------------------
// Kernel 4: SA2 MLP 131->128->128->256 + maxpool. One block / cloud, 8 waves.
// All three weight matrices resident in LDS, pre-swizzled to fragment order
// (139KB) + A/act staging; act2 aliases dead A buffer. ~280KB of 320KB LDS.
// ---------------------------------------------------------------------------
__global__ void __launch_bounds__(256)
k_sa2(const float* __restrict__ xyz, const float* __restrict__ f1,
      const float* __restrict__ nx2, const int* __restrict__ idx2,
      const float* __restrict__ w0, const float* __restrict__ b0,
      const float* __restrict__ w1, const float* __restrict__ b1,
      const float* __restrict__ w2, const float* __restrict__ b2,
      float* __restrict__ feats2) {
  extern __shared__ char smem[];
  half_t* wz0 = (half_t*)smem;          // KT=5,NT=8  : 40 tiles * 512
  half_t* wz1 = wz0 + 40 * 512;         // KT=4,NT=8  : 32 tiles * 512
  half_t* wz2 = wz1 + 32 * 512;         // KT=4,NT=16 : 64 tiles * 512
  half_t* a0  = wz2 + 64 * 512;         // 8 waves x [32][160]
  half_t* ac1 = a0 + 8 * 32 * 160;      // 8 waves x [32][128]

  const int b = blockIdx.x, t = threadIdx.x;
  const int wv = t >> 5, lane = t & 31;
  half_t* myA = a0 + wv * 32 * 160;
  half_t* my1 = ac1 + wv * 32 * 128;
  half_t* my2 = myA;                    // alias (ld 128), A dead after stage 1

  fill_b_swz(wz0, w0, 128, 131, 5, 8);  // [131,128] K-pad -> 160
  fill_b_swz(wz1, w1, 128, 128, 4, 8);  // [128,128]
  fill_b_swz(wz2, w2, 256, 128, 4, 16); // [128,256]
  __syncthreads();

  for (int ci = 0; ci < 2; ++ci) {
    const int c = wv + ci * 8;
    {  // gather A: row = lane, K = 160 (131 valid)
      const int j = idx2[((size_t)b * 16 + c) * 32 + lane];
      const float* pj = xyz + ((size_t)b * 32 + j) * 3;
      const float* pc = nx2 + ((size_t)b * 16 + c) * 3;
      half_t* r = myA + lane * 160;
      r[0] = (half_t)(pj[0] - pc[0]);
      r[1] = (half_t)(pj[1] - pc[1]);
      r[2] = (half_t)(pj[2] - pc[2]);
      const float* fj = f1 + ((size_t)b * 32 + j) * 128;
      for (int k = 0; k < 128; ++k) r[3 + k] = (half_t)fj[k];
#pragma unroll
      for (int k = 131; k < 160; ++k) r[k] = (half_t)0.f;
    }
    __syncthreads();

    // stage 1: [32,160(K=131)] @ w0[131,128]
#pragma unroll
    for (int mt = 0; mt < 2; ++mt)
      for (int nt = 0; nt < 8; ++nt) {
        v8f acc = {};
#pragma unroll
        for (int kt = 0; kt < 5; ++kt)
          acc = wmma32(load_a_lds(myA + mt * 16 * 160 + kt * 32, 160),
                       load_b_swz(wz0, kt * 8 + nt), acc);
        store_act(my1, 128, mt * 16, nt * 16, acc, b0);
      }
    __syncthreads();

    // stage 2: [32,128] @ w1[128,128]
#pragma unroll
    for (int mt = 0; mt < 2; ++mt)
      for (int nt = 0; nt < 8; ++nt) {
        v8f acc = {};
#pragma unroll
        for (int kt = 0; kt < 4; ++kt)
          acc = wmma32(load_a_lds(my1 + mt * 16 * 128 + kt * 32, 128),
                       load_b_swz(wz1, kt * 8 + nt), acc);
        store_act(my2, 128, mt * 16, nt * 16, acc, b1);
      }
    __syncthreads();

    // stage 3: [32,128] @ w2[128,256] fused with maxpool over samples
    float* outp = feats2 + ((size_t)b * 16 + c) * 256;
    for (int nt = 0; nt < 16; ++nt) {
      v8f acc0 = {}, acc1 = {};
#pragma unroll
      for (int kt = 0; kt < 4; ++kt) {
        v16h bb = load_b_swz(wz2, kt * 16 + nt);
        acc0 = wmma32(load_a_lds(my2 + kt * 32, 128), bb, acc0);
        acc1 = wmma32(load_a_lds(my2 + 16 * 128 + kt * 32, 128), bb, acc1);
      }
      maxpool32_store(outp, nt * 16, acc0, acc1, b2);
    }
    __syncthreads();
  }
}

// ---------------------------------------------------------------------------
// Kernel 5: SA3 MLP 259->256->512->768 + maxpool over 16 pts. 1 block / cloud.
// Single 16-row M tile; 8 waves split the N tiles; B streamed from global
// (weights too large for LDS; no intra-block reuse, L2-hot).
// ---------------------------------------------------------------------------
__global__ void __launch_bounds__(256)
k_sa3(const float* __restrict__ nx2, const float* __restrict__ f2,
      const float* __restrict__ w0, const float* __restrict__ b0,
      const float* __restrict__ w1, const float* __restrict__ b1,
      const float* __restrict__ w2, const float* __restrict__ b2,
      float* __restrict__ g) {
  __shared__ half_t A[16 * 288];     // K 259 -> 288
  __shared__ half_t ac1[16 * 256];
  __shared__ half_t ac2[16 * 512];
  const int b = blockIdx.x, t = threadIdx.x, wv = t >> 5;

  for (int i = t; i < 16 * 288; i += 256) {
    const int row = i / 288, k = i % 288;
    float v = 0.f;
    if (k < 3)        v = nx2[((size_t)b * 16 + row) * 3 + k];
    else if (k < 259) v = f2[((size_t)b * 16 + row) * 256 + (k - 3)];
    A[i] = (half_t)v;
  }
  __syncthreads();

  // stage 1: [16,288(K=259)] @ w0[259,256]
  for (int nt = wv; nt < 16; nt += 8) {
    v8f acc = {};
#pragma unroll
    for (int kt = 0; kt < 9; ++kt)
      acc = wmma32(load_a_lds(A + kt * 32, 288),
                   load_b_gmem(w0, 256, kt * 32, nt * 16, 259), acc);
    store_act(ac1, 256, 0, nt * 16, acc, b0);
  }
  __syncthreads();

  // stage 2: [16,256] @ w1[256,512]
  for (int nt = wv; nt < 32; nt += 8) {
    v8f acc = {};
#pragma unroll
    for (int kt = 0; kt < 8; ++kt)
      acc = wmma32(load_a_lds(ac1 + kt * 32, 256),
                   load_b_gmem(w1, 512, kt * 32, nt * 16, 256), acc);
    store_act(ac2, 512, 0, nt * 16, acc, b1);
  }
  __syncthreads();

  // stage 3: [16,512] @ w2[512,768] fused with maxpool over the 16 points
  float* outp = g + (size_t)b * 768;
  for (int nt = wv; nt < 48; nt += 8) {
    v8f acc = {};
#pragma unroll
    for (int kt = 0; kt < 16; ++kt)
      acc = wmma32(load_a_lds(ac2 + kt * 32, 512),
                   load_b_gmem(w2, 768, kt * 32, nt * 16, 512), acc);
    maxpool16_store(outp, nt * 16, acc, b2);
  }
}

// ---------------------------------------------------------------------------
// Kernel 6: batched FC as GEMM [B,768] @ [768,768] + bias. 4 waves / block;
// block = 16-row M tile x 64-col N chunk; A tile staged in LDS as f16 (paired
// stores), B streamed from global with prefetch.
// ---------------------------------------------------------------------------
__global__ void __launch_bounds__(128)
k_fc(const float* __restrict__ g, const float* __restrict__ W,
     const float* __restrict__ bias, float* __restrict__ out) {
  __shared__ half_t As[16 * 768];
  const int t = threadIdx.x, wv = t >> 5, lane = t & 31;
  const int m0 = blockIdx.x * 16;
  const int n0 = blockIdx.y * 64 + wv * 16;

  for (int i = t; i < 16 * 768 / 2; i += 128) {
    const float* gp = g + (size_t)m0 * 768 + (size_t)i * 2;
    v2h h; h[0] = (half_t)gp[0]; h[1] = (half_t)gp[1];
    ((v2h*)As)[i] = h;
  }
  __syncthreads();

  v8f acc = {};
  for (int kt = 0; kt < 24; ++kt)
    acc = wmma32(load_a_lds(As + kt * 32, 768),
                 load_b_gmem(W, 768, kt * 32, n0, 768), acc);

  const int col = n0 + (lane & 15);
  const int r0  = m0 + ((lane & 16) ? 8 : 0);
  const float bv = bias[col];
#pragma unroll
  for (int v = 0; v < 8; ++v)
    out[(size_t)(r0 + v) * 768 + col] = acc[v] + bv;
}

// ---------------------------------------------------------------------------
// Host launcher
// ---------------------------------------------------------------------------
extern "C" void kernel_launch(void* const* d_in, const int* in_sizes, int n_in,
                              void* d_out, int out_size, void* d_ws, size_t ws_size,
                              hipStream_t stream) {
  const float* x    = (const float*)d_in[0];
  const float* w1_0 = (const float*)d_in[1];  const float* b1_0 = (const float*)d_in[2];
  const float* w1_1 = (const float*)d_in[3];  const float* b1_1 = (const float*)d_in[4];
  const float* w1_2 = (const float*)d_in[5];  const float* b1_2 = (const float*)d_in[6];
  const float* w2_0 = (const float*)d_in[7];  const float* b2_0 = (const float*)d_in[8];
  const float* w2_1 = (const float*)d_in[9];  const float* b2_1 = (const float*)d_in[10];
  const float* w2_2 = (const float*)d_in[11]; const float* b2_2 = (const float*)d_in[12];
  const float* w3_0 = (const float*)d_in[13]; const float* b3_0 = (const float*)d_in[14];
  const float* w3_1 = (const float*)d_in[15]; const float* b3_1 = (const float*)d_in[16];
  const float* w3_2 = (const float*)d_in[17]; const float* b3_2 = (const float*)d_in[18];
  const float* fc_w = (const float*)d_in[19]; const float* fc_b = (const float*)d_in[20];

  const int B = in_sizes[0] / (NPTS * NCH);   // 2048

  // Workspace carve-out (~87 MB total)
  char* ws = (char*)d_ws;
  int*   idx1 = (int*)ws;   ws += (size_t)B * 32 * 32 * sizeof(int);
  float* nx1  = (float*)ws; ws += (size_t)B * 32 * 3 * sizeof(float);
  float* f1   = (float*)ws; ws += (size_t)B * 32 * 128 * sizeof(float);
  int*   idx2 = (int*)ws;   ws += (size_t)B * 16 * 32 * sizeof(int);
  float* nx2  = (float*)ws; ws += (size_t)B * 16 * 3 * sizeof(float);
  float* f2   = (float*)ws; ws += (size_t)B * 16 * 256 * sizeof(float);
  float* gbuf = (float*)ws; ws += (size_t)B * 768 * sizeof(float);
  (void)ws_size; (void)n_in; (void)out_size;

  const int smem_sa1 = (4 * 512 + 8 * 512 + 16 * 512 +
                        8 * 32 * 32 + 8 * 32 * 64 + 8 * 32 * 64) * (int)sizeof(half_t);
  const int smem_sa2 = ((40 + 32 + 64) * 512 +
                        8 * 32 * 160 + 8 * 32 * 128) * (int)sizeof(half_t);
  (void)hipFuncSetAttribute(reinterpret_cast<const void*>(k_sa1),
                            hipFuncAttributeMaxDynamicSharedMemorySize, smem_sa1);
  (void)hipFuncSetAttribute(reinterpret_cast<const void*>(k_sa2),
                            hipFuncAttributeMaxDynamicSharedMemorySize, smem_sa2);

  k_fps_bq1<<<B, 256, 0, stream>>>(x, idx1, nx1);
  k_sa1<<<B, 256, smem_sa1, stream>>>(x, nx1, idx1,
                                      w1_0, b1_0, w1_1, b1_1, w1_2, b1_2, f1);
  k_fps_bq2<<<B, 32, 0, stream>>>(nx1, idx2, nx2);
  k_sa2<<<B, 256, smem_sa2, stream>>>(nx1, f1, nx2, idx2,
                                      w2_0, b2_0, w2_1, b2_1, w2_2, b2_2, f2);
  k_sa3<<<B, 256, 0, stream>>>(nx2, f2,
                               w3_0, b3_0, w3_1, b3_1, w3_2, b3_2, gbuf);
  dim3 gfc(B / 16, 768 / 64);
  k_fc<<<gfc, 128, 0, stream>>>(gbuf, fc_w, fc_b, (float*)d_out);
}
